// AttentionLayer_55379308315424
// MI455X (gfx1250) — matrix operand order repaired and checked by reading, not compiled, not run on hardware
//
#include <hip/hip_runtime.h>

// ---------------------------------------------------------------------------
// AttentionLayer on gfx1250 (MI455X), wave32, f16 WMMA with f32 accumulate.
// Shapes: B=8, C=768, CK=96, H=W=64, HW=4096, L=CK*H=6144.
// q/k/v/attn stored [96,4096] row-major == [6144,64] row-major (same memory).
// Pipeline:
//   k0a: Wq/Wk/Wv/Wc f32 -> f16 (once)
//   k0b: x [C,HW] f32 -> xT16 [HW,C] f16 (tiled LDS transpose, coalesced)
//   k1 : q/k/v projections, all-contiguous f16 fragment loads
//   k2 : fused flash attention (online softmax, LDS P re-swizzle)
//   k3 : out = gamma * (Wc @ attn) + x
// ---------------------------------------------------------------------------

typedef __attribute__((ext_vector_type(16))) _Float16 v16h;
typedef __attribute__((ext_vector_type(8)))  _Float16 v8h;
typedef __attribute__((ext_vector_type(8)))  float    v8f;

#define DEVINL __device__ __forceinline__

static constexpr int kB  = 8;
static constexpr int kC  = 768;
static constexpr int kCK = 96;
static constexpr int kHW = 4096;
static constexpr int kL  = 6144;   // CK * H
static constexpr int kW  = 64;

DEVINL v8f wmma_f16(v16h a, v16h b, v8f c) {
  // D = A(16x32 f16) x B(32x16 f16) + C(16x16 f32)
  return __builtin_amdgcn_wmma_f32_16x16x32_f16(false, a, false, b, (short)0, c,
                                                false, false);
}

// --- A fragment (16x32): lane -> M=lane%16, element e ->
//     K = (lane/16)*16 + e.  Source f16 row-major, ld halfs/row.
DEVINL v16h load_a_f16(const _Float16* base, int ld, int lane) {
  const int m = lane & 15, g = lane >> 4;
  const _Float16* p = base + (size_t)m * ld + g * 16;
  v8h lo = *(const v8h*)(p);
  v8h hi = *(const v8h*)(p + 8);
  v16h a;
#pragma unroll
  for (int e = 0; e < 8; ++e) { a[e] = lo[e]; a[8 + e] = hi[e]; }
  return a;
}

// --- B fragment (32x16): lane -> N=lane%16, element e -> K = (lane/16)*16 + e.
// Source stores, for each n, 32 contiguous k halfs: element(k,n)=base[n*stride+k]
DEVINL v16h load_b_f16_kcontig(const _Float16* base, int stride, int lane) {
  const int n = lane & 15, g = lane >> 4;
  return *(const v16h*)(base + (size_t)n * stride + (size_t)g * 16);
}

// ---------------------------------------------------------------------------
// Kernel 0a: plain f32 -> f16 conversion (weights; tiny, launched per tensor).
// ---------------------------------------------------------------------------
__global__ __launch_bounds__(256) void cvt_f32_f16_kernel(
    const float* __restrict__ s, _Float16* __restrict__ d, int n) {
  const int i = blockIdx.x * blockDim.x + threadIdx.x;
  if (i < n) d[i] = (_Float16)s[i];
}

// ---------------------------------------------------------------------------
// Kernel 0b: x [768,4096] f32 -> xT16 [4096,768] f16 per batch.
// 64x64 tile per block through LDS; reads and writes fully coalesced.
// ---------------------------------------------------------------------------
__global__ __launch_bounds__(256) void transpose_x_kernel(
    const float* __restrict__ x, _Float16* __restrict__ xT) {
  __shared__ _Float16 tile[64][72];            // pad to dodge bank conflicts

  const int bid = blockIdx.x;
  const int hwt = bid & 63;                    // 4096/64
  const int ct  = (bid >> 6) % 12;             // 768/64
  const int b   = bid / (64 * 12);
  const int c0 = ct * 64, hw0 = hwt * 64;
  const int t = threadIdx.x;

  {  // load: thread covers 16 consecutive hw of one c row
    const int cl = t >> 2, chw = (t & 3) * 16;
    const float* p = x + (size_t)b * kC * kHW + (size_t)(c0 + cl) * kHW + hw0 + chw;
#pragma unroll
    for (int e = 0; e < 16; ++e) tile[cl][chw + e] = (_Float16)p[e];
  }
  __syncthreads();
  {  // store: thread writes 16 consecutive c of one hw row (one 32B store)
    const int hwl = t >> 2, cc = (t & 3) * 16;
    v16h v;
#pragma unroll
    for (int e = 0; e < 16; ++e) v[e] = tile[cc + e][hwl];
    _Float16* q = xT + (size_t)b * kHW * kC + (size_t)(hw0 + hwl) * kC + c0 + cc;
    *(v16h*)q = v;
  }
}

// ---------------------------------------------------------------------------
// Kernel 1: q/k/v projections.  One wave = one (b, mtile of 96, ntile of 4096);
// the xT B-fragment is shared by the three weight A-fragments (3 WMMA / chunk).
// q,k stored f16 row-major [96,4096] (== [6144,64]); v stored transposed
// vT[w=0..63][l=0..6143] so attention's V B-fragments are contiguous loads.
// ---------------------------------------------------------------------------
__global__ __launch_bounds__(256) void qkv_proj_kernel(
    const _Float16* __restrict__ xT, const _Float16* __restrict__ Wq,
    const _Float16* __restrict__ Wk, const _Float16* __restrict__ Wv,
    _Float16* __restrict__ q16, _Float16* __restrict__ k16,
    _Float16* __restrict__ vT16) {
  const int lane = threadIdx.x & 31;
  const int gw = (blockIdx.x * blockDim.x + threadIdx.x) >> 5;
  const int nt = gw & 255;           // 256 n-tiles
  const int mt = (gw >> 8) % 6;      // 6 m-tiles (96/16)
  const int b  = gw / (256 * 6);
  const int m0 = mt * 16, n0 = nt * 16;
  const _Float16* xb = xT + (size_t)b * kHW * kC;

  v8f cq = {}, ck = {}, cv = {};
  for (int kk = 0; kk < kC; kk += 32) {
    // element(k,n) = xT[(n0+n)*768 + kk + k] : contiguous 32B per lane
    v16h bx = load_b_f16_kcontig(xb + (size_t)n0 * kC + kk, kC, lane);
    v16h aq = load_a_f16(Wq + (size_t)m0 * kC + kk, kC, lane);
    cq = wmma_f16(aq, bx, cq);
    v16h ak = load_a_f16(Wk + (size_t)m0 * kC + kk, kC, lane);
    ck = wmma_f16(ak, bx, ck);
    v16h av = load_a_f16(Wv + (size_t)m0 * kC + kk, kC, lane);
    cv = wmma_f16(av, bx, cv);
  }

  // D layout: VGPR r holds row M = r + 8*(lane/16); N = lane%16.
  const int g = lane >> 4, n = lane & 15;
  const size_t base = (size_t)b * kL * kW;
  const int hw = n0 + n;
  const int h = hw >> 6, w = hw & 63;
#pragma unroll
  for (int r = 0; r < 8; ++r) {
    const int o = m0 + r + 8 * g;                       // output channel
    q16[base + (size_t)o * kHW + hw] = (_Float16)cq[r];
    k16[base + (size_t)o * kHW + hw] = (_Float16)ck[r];
    vT16[base + (size_t)w * kL + o * kW + h] = (_Float16)cv[r];  // vT[w][l]
  }
}

// ---------------------------------------------------------------------------
// Kernel 2: fused flash attention. One wave owns 16 q-rows (of 6144); block has
// 8 waves. Online softmax with stats kept in the C/D row distribution (row m in
// VGPR m%8 of lane-group m/8); P is re-swizzled to the A-fragment layout via a
// per-wave 16x32 LDS staging buffer. Output written transposed [4096,96] f16.
// ---------------------------------------------------------------------------
__global__ __launch_bounds__(256) void flash_attn_kernel(
    const _Float16* __restrict__ q16, const _Float16* __restrict__ k16,
    const _Float16* __restrict__ vT16, _Float16* __restrict__ aT16) {
  __shared__ _Float16 Pbuf[8][16][32];   // per-wave P staging (8 KB)

  const int lane = threadIdx.x & 31;
  const int wid  = threadIdx.x >> 5;
  const int gw   = blockIdx.x * 8 + wid;
  const int b    = gw / 384;             // 384 row-tiles per batch
  const int rt   = gw % 384;
  const int l0   = rt * 16;
  const int g = lane >> 4, n = lane & 15;

  const size_t base = (size_t)b * kL * kW;
  const _Float16* qb = q16 + base;
  const _Float16* kb = k16 + base;
  const _Float16* vb = vT16 + base;

  // Q A-fragments for this wave's 16 rows (K = w, 0..63 -> two fragments).
  const v16h aq0 = load_a_f16(qb + (size_t)l0 * kW + 0,  kW, lane);
  const v16h aq1 = load_a_f16(qb + (size_t)l0 * kW + 32, kW, lane);

  float mr[8], lr[8];
  v8f oacc[4] = {{}, {}, {}, {}};
#pragma unroll
  for (int r = 0; r < 8; ++r) { mr[r] = -1e30f; lr[r] = 0.0f; }

  for (int ch = 0; ch < kL; ch += 32) {   // 192 column chunks of 32
    // ---- S tiles: rows l0..l0+15, columns ch..ch+31 (two 16x16 tiles) ----
    v8f c0 = {}, c1 = {};
    {
      v16h bk = load_b_f16_kcontig(kb + (size_t)ch * kW + 0,  kW, lane);
      c0 = wmma_f16(aq0, bk, c0);
      bk = load_b_f16_kcontig(kb + (size_t)ch * kW + 32, kW, lane);
      c0 = wmma_f16(aq1, bk, c0);
      bk = load_b_f16_kcontig(kb + (size_t)(ch + 16) * kW + 0,  kW, lane);
      c1 = wmma_f16(aq0, bk, c1);
      bk = load_b_f16_kcontig(kb + (size_t)(ch + 16) * kW + 32, kW, lane);
      c1 = wmma_f16(aq1, bk, c1);
    }

    // ---- online softmax over the 32 new columns ----
#pragma unroll
    for (int r = 0; r < 8; ++r) {
      float v0 = c0[r], v1 = c1[r];
      float mx = fmaxf(v0, v1);
#pragma unroll
      for (int d = 1; d < 16; d <<= 1) mx = fmaxf(mx, __shfl_xor(mx, d, 32));
      const float mnew = fmaxf(mr[r], mx);
      const float sc = __expf(mr[r] - mnew);
      const float p0 = __expf(v0 - mnew);
      const float p1 = __expf(v1 - mnew);
      float sum = p0 + p1;
#pragma unroll
      for (int d = 1; d < 16; d <<= 1) sum += __shfl_xor(sum, d, 32);
      lr[r] = lr[r] * sc + sum;
      mr[r] = mnew;
#pragma unroll
      for (int wt = 0; wt < 4; ++wt) oacc[wt][r] *= sc;
      // stage P in row-major [row][col] (rows = q rows, cols = 32 l's of chunk)
      Pbuf[wid][r + 8 * g][n]      = (_Float16)p0;
      Pbuf[wid][r + 8 * g][16 + n] = (_Float16)p1;
    }
    asm volatile("s_wait_dscnt 0" ::: "memory");  // wave-private LDS RAW

    // ---- O += P(16x32) @ V(32x64) ----
    const v16h ap = load_a_f16(&Pbuf[wid][0][0], 32, lane);
#pragma unroll
    for (int wt = 0; wt < 4; ++wt) {
      v16h bv = load_b_f16_kcontig(vb + (size_t)(wt * 16) * kL + ch, kL, lane);
      oacc[wt] = wmma_f16(ap, bv, oacc[wt]);
    }
  }

  // ---- finalize and store attn transposed: aT[hw][o], hw = h*64+w ----
  const int o  = l0 >> 6;        // l = o*64 + h; 16-row tile stays in one o
  const int h0 = l0 & 63;
#pragma unroll
  for (int r = 0; r < 8; ++r) {
    const float inv = 1.0f / lr[r];
    const int h = h0 + r + 8 * g;
#pragma unroll
    for (int wt = 0; wt < 4; ++wt) {
      const int w = wt * 16 + n;
      aT16[base + (size_t)(h * 64 + w) * kCK + o] = (_Float16)(oacc[wt][r] * inv);
    }
  }
}

// ---------------------------------------------------------------------------
// Kernel 3: out = gamma * (Wc @ attn) + x.  One wave = one 16x16 tile of
// [768, 4096]; K = 96 contracted in 3 chunks of 32 from transposed attn.
// ---------------------------------------------------------------------------
__global__ __launch_bounds__(256) void out_proj_kernel(
    const float* __restrict__ x, const _Float16* __restrict__ Wc,
    const float* __restrict__ gamma, const _Float16* __restrict__ aT16,
    float* __restrict__ out) {
  const int lane = threadIdx.x & 31;
  const int gw = (blockIdx.x * blockDim.x + threadIdx.x) >> 5;
  const int nt = gw & 255;            // 256 n-tiles
  const int mt = (gw >> 8) % 48;      // 48 m-tiles (768/16)
  const int b  = gw / (256 * 48);
  const int m0 = mt * 16, n0 = nt * 16;
  const _Float16* ab = aT16 + (size_t)b * kHW * kCK;

  v8f acc = {};
#pragma unroll
  for (int kk = 0; kk < kCK; kk += 32) {
    v16h a = load_a_f16(Wc + (size_t)m0 * kCK + kk, kCK, lane);
    v16h bf = load_b_f16_kcontig(ab + (size_t)n0 * kCK + kk, kCK, lane);
    acc = wmma_f16(a, bf, acc);
  }

  const float gm = gamma[0];
  const int g = lane >> 4, n = lane & 15;
  const int hw = n0 + n;
#pragma unroll
  for (int r = 0; r < 8; ++r) {
    const int c = m0 + r + 8 * g;
    const size_t idx = (size_t)b * kC * kHW + (size_t)c * kHW + hw;
    out[idx] = gm * acc[r] + x[idx];
  }
}

// ---------------------------------------------------------------------------
extern "C" void kernel_launch(void* const* d_in, const int* in_sizes, int n_in,
                              void* d_out, int out_size, void* d_ws, size_t ws_size,
                              hipStream_t stream) {
  const float* x     = (const float*)d_in[0];
  const float* Wq    = (const float*)d_in[1];
  const float* Wk    = (const float*)d_in[2];
  const float* Wv    = (const float*)d_in[3];
  const float* Wc    = (const float*)d_in[4];
  const float* gamma = (const float*)d_in[5];
  float* out = (float*)d_out;

  const size_t per   = (size_t)kB * kL * kW;   // 3,145,728 halfs per tensor
  const size_t perXT = (size_t)kB * kHW * kC;  // 25,165,824 halfs
  const size_t wN    = (size_t)kCK * kC;       // 73,728 halfs per weight

  _Float16* q16  = (_Float16*)d_ws;
  _Float16* k16  = q16 + per;
  _Float16* vT   = k16 + per;
  _Float16* aT   = vT + per;
  _Float16* xT   = aT + per;
  _Float16* Wq16 = xT + perXT;
  _Float16* Wk16 = Wq16 + wN;
  _Float16* Wv16 = Wk16 + wN;
  _Float16* Wc16 = Wv16 + wN;                  // total ~76 MB of d_ws

  // k0a: weight conversions (tiny)
  cvt_f32_f16_kernel<<<(int)((wN + 255) / 256), 256, 0, stream>>>(Wq, Wq16, (int)wN);
  cvt_f32_f16_kernel<<<(int)((wN + 255) / 256), 256, 0, stream>>>(Wk, Wk16, (int)wN);
  cvt_f32_f16_kernel<<<(int)((wN + 255) / 256), 256, 0, stream>>>(Wv, Wv16, (int)wN);
  cvt_f32_f16_kernel<<<(int)((wN + 255) / 256), 256, 0, stream>>>(Wc, Wc16, (int)wN);
  // k0b: x transpose+convert: 8 * 12 * 64 = 6144 blocks
  transpose_x_kernel<<<6144, 256, 0, stream>>>(x, xT);
  // k1: 8 batches * 6 m-tiles * 256 n-tiles = 12288 waves / 8 per block
  qkv_proj_kernel<<<1536, 256, 0, stream>>>(xT, Wq16, Wk16, Wv16, q16, k16, vT);
  // k2: 8 batches * 384 row-tiles = 3072 waves / 8 per block
  flash_attn_kernel<<<384, 256, 0, stream>>>(q16, k16, vT, aT);
  // k3: 8 * 48 * 256 = 98304 waves / 8 per block
  out_proj_kernel<<<12288, 256, 0, stream>>>(x, Wc16, gamma, aT, out);
}
